// GAT_23931557773762
// MI455X (gfx1250) — matrix operand order repaired and checked by reading, not compiled, not run on hardware
//
#include <hip/hip_runtime.h>
#include <stdint.h>

// ---------------------------------------------------------------------------
// GAT forward for MI455X (gfx1250, wave32).
// Dense GEMMs -> v_wmma_f32_16x16x32_f16 (f16 A/B, f32 accum).
// Edge softmax/aggregation -> L2-resident atomics (whole working set < 192MB L2).
// ---------------------------------------------------------------------------

typedef __attribute__((ext_vector_type(16))) _Float16 v16h;
typedef __attribute__((ext_vector_type(8)))  _Float16 h8;
typedef __attribute__((ext_vector_type(8)))  float    v8f;

#define NEG_SLOPE 0.2f

// ---------------- float atomic max via sign-split integer atomics -----------
__device__ __forceinline__ void atomicMaxF(float* addr, float v) {
    if (v >= 0.0f) atomicMax((int*)addr, __float_as_int(v));
    else           atomicMin((unsigned int*)addr, __float_as_uint(v));
}

// ---------------- f32 -> f16 convert ---------------------------------------
__global__ void k_f32_to_f16(const float* __restrict__ in, _Float16* __restrict__ out,
                             long long total) {
    long long t = (long long)blockIdx.x * blockDim.x + threadIdx.x;
    if (t < total) out[t] = (_Float16)in[t];
}

__global__ void k_zero_f32(float* __restrict__ p, long long total) {
    long long t = (long long)blockIdx.x * blockDim.x + threadIdx.x;
    if (t < total) p[t] = 0.0f;
}

// ---------------- pack weight matrix into WMMA B-fragment order -------------
// B operand order used here (16x16x32 f16): lane l holds column N = l%16; half i holds
// K = 16*(l/16) + i  (lanes 0-15: K=0..15, lanes 16-31: K=16..31).
// Packed layout: Bp[((kt*NT + nt)*32 + lane)*16 + i]
__global__ void k_pack_b(const float* __restrict__ W, _Float16* __restrict__ Bp,
                         int K, int Ncols) {
    int KT = K >> 5, NT = Ncols >> 4;
    int t = blockIdx.x * blockDim.x + threadIdx.x;
    if (t >= KT * NT * 32) return;
    int lane = t & 31;
    int nt   = (t >> 5) % NT;
    int kt   = (t >> 5) / NT;
    int col  = lane & 15;
    int hi   = lane >> 4;
    #pragma unroll
    for (int i = 0; i < 16; ++i) {
        int kk = kt * 32 + hi * 16 + i;
        Bp[(size_t)t * 16 + i] = (_Float16)W[(size_t)kk * Ncols + nt * 16 + col];
    }
}

// ---------------- WMMA GEMM: C[M,Ncols] = A[M,K] * B[K,Ncols] ---------------
// One wave per 16-row stripe; NT column tiles per wave (A fragment reused NT x).
// A fragment (16x32 f16): lane l holds row M=l%16; K order matches packed B:
//   lanes 0-15 : halfs 0-15 = K 0..15
//   lanes 16-31: halfs 0-15 = K 16..31
// -> two contiguous 16B loads at K-offset (hi*16) and (hi*16 + 8).
template <int NT>
__global__ void k_gemm_wmma(const _Float16* __restrict__ A,
                            const _Float16* __restrict__ Bp,
                            float* __restrict__ C,
                            int Mtiles, int K, int Ncols) {
    int wave = (int)((blockIdx.x * blockDim.x + threadIdx.x) >> 5);
    int lane = threadIdx.x & 31;
    if (wave >= Mtiles) return;          // wave-uniform: EXEC all-1s at WMMA
    int m16 = lane & 15;
    int hi  = lane >> 4;
    int row = wave * 16 + m16;

    v8f acc[NT] = {};
    int KT = K >> 5;
    for (int kt = 0; kt < KT; ++kt) {
        int off0 = kt * 32 + (hi << 4);
        h8 lo = *(const h8*)(A + (size_t)row * K + off0);
        h8 hh = *(const h8*)(A + (size_t)row * K + off0 + 8);
        v16h a;
        #pragma unroll
        for (int i = 0; i < 8; ++i) { a[i] = lo[i]; a[i + 8] = hh[i]; }
        #pragma unroll
        for (int nt = 0; nt < NT; ++nt) {
            v16h b = *(const v16h*)(Bp + ((size_t)(kt * NT + nt) * 32 + lane) * 16);
            acc[nt] = __builtin_amdgcn_wmma_f32_16x16x32_f16(
                false, a, false, b, (short)0, acc[nt], false, false);
        }
    }
    // C/D layout: lanes 0-15 VGPR r -> M=r; lanes 16-31 VGPR r -> M=r+8; N=lane%16
    int baseRow = wave * 16 + hi * 8;
    #pragma unroll
    for (int nt = 0; nt < NT; ++nt)
        #pragma unroll
        for (int r = 0; r < 8; ++r)
            C[(size_t)(baseRow + r) * Ncols + nt * 16 + m16] = acc[nt][r];
}

// ---------------- per-node attention scores + state init --------------------
__global__ void k_node_prep(const float* __restrict__ h,
                            const float* __restrict__ a_s, const float* __restrict__ a_d,
                            float* __restrict__ as_, float* __restrict__ ad_,
                            float* __restrict__ m, float* __restrict__ den,
                            float* __restrict__ deg, int N, int initDeg) {
    int n = blockIdx.x * blockDim.x + threadIdx.x;
    if (n >= N) return;
    const float* hn = h + (size_t)n * 96;
    float s = 0.f, d = 0.f;
    #pragma unroll 8
    for (int i = 0; i < 96; ++i) { s += hn[i] * a_s[i]; d += hn[i] * a_d[i]; }
    as_[n] = s; ad_[n] = d;
    m[n]   = -3.402823466e38f;
    den[n] = 0.f;
    if (initDeg) deg[n] = 0.f;
}

// ---------------- edge pass 1: logits + segment max (+deg) ------------------
__global__ void k_edge_logits(const long long* __restrict__ ei,
                              const float* __restrict__ as_, const float* __restrict__ ad_,
                              float* __restrict__ ee, float* __restrict__ m,
                              float* __restrict__ deg, long long E, int N, int addDeg) {
    long long e = (long long)blockIdx.x * blockDim.x + threadIdx.x;
    if (e >= E + N) return;
    int s, d;
    if (e < E) { s = (int)ei[e]; d = (int)ei[E + e]; }
    else       { s = d = (int)(e - E); }               // self loop
    float v = as_[s] + ad_[d];
    v = v > 0.f ? v : NEG_SLOPE * v;                   // leaky_relu
    ee[e] = v;
    atomicMaxF(m + d, v);
    if (addDeg) atomicAdd(deg + d, 1.0f);
}

// ---------------- edge pass 2: softmax numerator + weighted scatter ---------
// 24 threads per edge; each handles 4 of 96 channels (float4 gather).
__global__ void k_edge_scatter(const long long* __restrict__ ei,
                               const float* __restrict__ ee, const float* __restrict__ m,
                               const float* __restrict__ h,
                               float* __restrict__ den, float* __restrict__ agg,
                               long long E, int N) {
    long long t = (long long)blockIdx.x * blockDim.x + threadIdx.x;
    if (t >= (E + N) * 24) return;
    long long e = t / 24;
    int c4 = (int)(t % 24) * 4;
    int s, d;
    if (e < E) { s = (int)ei[e]; d = (int)ei[E + e]; }
    else       { s = d = (int)(e - E); }
    float p = __expf(ee[e] - m[d]);
    if (c4 == 0) atomicAdd(den + d, p);
    float4 hv = *(const float4*)(h + (size_t)s * 96 + c4);
    float* ag = agg + (size_t)d * 96 + c4;
    atomicAdd(ag + 0, p * hv.x);
    atomicAdd(ag + 1, p * hv.y);
    atomicAdd(ag + 2, p * hv.z);
    atomicAdd(ag + 3, p * hv.w);
}

// ---------------- GAT finalize + faithful flat reshape(3,N,32).sum(0) -------
// val_flat[f] = agg[f]/(max(den,1e-16)*max(deg,1))[f/96] + bias[f%96]
// out[n,j]    = (relu?) sum_g val_flat[g*N*32 + n*32 + j]   -> f16
__global__ void k_finalize(const float* __restrict__ agg, const float* __restrict__ den,
                           const float* __restrict__ deg, const float* __restrict__ bias,
                           _Float16* __restrict__ outh, int N, int doRelu) {
    int t = blockIdx.x * blockDim.x + threadIdx.x;
    if (t >= N * 32) return;
    float s = 0.f;
    #pragma unroll
    for (int g = 0; g < 3; ++g) {
        size_t f   = (size_t)g * N * 32 + t;
        int    row = (int)(f / 96);
        int    col = (int)(f % 96);
        float inv = 1.f / (fmaxf(den[row], 1e-16f) * fmaxf(deg[row], 1.f));
        s += agg[f] * inv + bias[col];
    }
    if (doRelu) s = fmaxf(s, 0.f);
    outh[t] = (_Float16)s;
}

// ---------------- bias + relu + f16 convert (lin1) --------------------------
__global__ void k_bias_relu_h(const float* __restrict__ in, const float* __restrict__ bias,
                              _Float16* __restrict__ outh, long long total, int C) {
    long long t = (long long)blockIdx.x * blockDim.x + threadIdx.x;
    if (t >= total) return;
    float v = in[t] + bias[t % C];
    outh[t] = (_Float16)fmaxf(v, 0.f);
}

// ---------------- bias + sigmoid (in place on d_out) ------------------------
__global__ void k_sigmoid_bias(float* __restrict__ io, const float* __restrict__ bias,
                               long long total, int C) {
    long long t = (long long)blockIdx.x * blockDim.x + threadIdx.x;
    if (t >= total) return;
    float v = io[t] + bias[t % C];
    io[t] = 1.0f / (1.0f + __expf(-v));
}

// ---------------------------------------------------------------------------
extern "C" void kernel_launch(void* const* d_in, const int* in_sizes, int n_in,
                              void* d_out, int out_size, void* d_ws, size_t ws_size,
                              hipStream_t stream) {
    const float*     x    = (const float*)d_in[0];
    const long long* ei   = (const long long*)d_in[1];   // int64 edge_index [2,E]
    const float*     W1   = (const float*)d_in[2];
    const float*     as1  = (const float*)d_in[3];
    const float*     ad1  = (const float*)d_in[4];
    const float*     b1   = (const float*)d_in[5];
    const float*     W2   = (const float*)d_in[6];
    const float*     as2  = (const float*)d_in[7];
    const float*     ad2  = (const float*)d_in[8];
    const float*     b2   = (const float*)d_in[9];
    // d_in[10..11] = W_skip/b_skip: multiplied by 0.0 in reference -> skipped
    const float*     Wl1  = (const float*)d_in[12];
    const float*     bl1  = (const float*)d_in[13];
    const float*     Wl2  = (const float*)d_in[14];
    const float*     bl2  = (const float*)d_in[15];

    const int       N  = in_sizes[0] / 128;     // 100000 (multiple of 16)
    const long long E  = in_sizes[1] / 2;       // 1.6M
    const int  Mtiles  = N / 16;

    // ---- workspace carve-out (256B aligned) ----
    uint8_t* base = (uint8_t*)d_ws;
    size_t off = 0;
    auto alloc = [&](size_t bytes) -> size_t {
        size_t o = off; off = (off + bytes + 255) & ~(size_t)255; return o;
    };
    _Float16* Axh = (_Float16*)(base + alloc((size_t)N * 128 * 2)); // x in f16
    float*    Hf  = (float*)   (base + alloc((size_t)N * 96 * 4));  // GEMM outputs
    float*    AGG = (float*)   (base + alloc((size_t)N * 96 * 4));  // scatter accum
    float*    EE  = (float*)   (base + alloc((size_t)(E + N) * 4)); // edge logits
    float*    AS  = (float*)   (base + alloc((size_t)N * 4));
    float*    AD  = (float*)   (base + alloc((size_t)N * 4));
    float*    MX  = (float*)   (base + alloc((size_t)N * 4));
    float*    DEN = (float*)   (base + alloc((size_t)N * 4));
    float*    DEG = (float*)   (base + alloc((size_t)N * 4));
    _Float16* Fh  = (_Float16*)(base + alloc((size_t)N * 96 * 2));  // f16 staging
    _Float16* PW1 = (_Float16*)(base + alloc(4 * 6 * 32 * 16 * 2));
    _Float16* PW2 = (_Float16*)(base + alloc(1 * 6 * 32 * 16 * 2));
    _Float16* PW3 = (_Float16*)(base + alloc(1 * 6 * 32 * 16 * 2));
    _Float16* PW4 = (_Float16*)(base + alloc(3 * 2 * 32 * 16 * 2));

    const int B = 256;
    auto blk = [](long long tot, int b) { return (unsigned)((tot + b - 1) / b); };
    const long long TOT_E = E + N;

    // ---- stage 0: conversions + weight packing ----
    k_f32_to_f16<<<blk((long long)N * 128, B), B, 0, stream>>>(x, Axh, (long long)N * 128);
    k_pack_b<<<blk(4 * 6 * 32, B), B, 0, stream>>>(W1,  PW1, 128, 96);
    k_pack_b<<<blk(1 * 6 * 32, B), B, 0, stream>>>(W2,  PW2,  32, 96);
    k_pack_b<<<blk(1 * 6 * 32, B), B, 0, stream>>>(Wl1, PW3,  32, 96);
    k_pack_b<<<blk(3 * 2 * 32, B), B, 0, stream>>>(Wl2, PW4,  96, 32);

    // ---- GAT layer 1 ----
    k_gemm_wmma<6><<<blk((long long)Mtiles * 32, B), B, 0, stream>>>(Axh, PW1, Hf, Mtiles, 128, 96);
    k_node_prep<<<blk(N, B), B, 0, stream>>>(Hf, as1, ad1, AS, AD, MX, DEN, DEG, N, 1);
    k_zero_f32<<<blk((long long)N * 96, B), B, 0, stream>>>(AGG, (long long)N * 96);
    k_edge_logits<<<blk(TOT_E, B), B, 0, stream>>>(ei, AS, AD, EE, MX, DEG, E, N, 1);
    k_edge_scatter<<<blk(TOT_E * 24, B), B, 0, stream>>>(ei, EE, MX, Hf, DEN, AGG, E, N);
    k_finalize<<<blk((long long)N * 32, B), B, 0, stream>>>(AGG, DEN, DEG, b1, Fh, N, 1);

    // ---- GAT layer 2 (input: Fh [N,32] f16) ----
    k_gemm_wmma<6><<<blk((long long)Mtiles * 32, B), B, 0, stream>>>(Fh, PW2, Hf, Mtiles, 32, 96);
    k_node_prep<<<blk(N, B), B, 0, stream>>>(Hf, as2, ad2, AS, AD, MX, DEN, DEG, N, 0);
    k_zero_f32<<<blk((long long)N * 96, B), B, 0, stream>>>(AGG, (long long)N * 96);
    k_edge_logits<<<blk(TOT_E, B), B, 0, stream>>>(ei, AS, AD, EE, MX, DEG, E, N, 0);
    k_edge_scatter<<<blk(TOT_E * 24, B), B, 0, stream>>>(ei, EE, MX, Hf, DEN, AGG, E, N);
    k_finalize<<<blk((long long)N * 32, B), B, 0, stream>>>(AGG, DEN, DEG, b2, Fh, N, 0);

    // ---- MLP head ----
    k_gemm_wmma<6><<<blk((long long)Mtiles * 32, B), B, 0, stream>>>(Fh, PW3, Hf, Mtiles, 32, 96);
    k_bias_relu_h<<<blk((long long)N * 96, B), B, 0, stream>>>(Hf, bl1, Fh, (long long)N * 96, 96);
    k_gemm_wmma<2><<<blk((long long)Mtiles * 32, B), B, 0, stream>>>(Fh, PW4, (float*)d_out, Mtiles, 96, 32);
    k_sigmoid_bias<<<blk((long long)N * 32, B), B, 0, stream>>>((float*)d_out, bl2, (long long)N * 32, 32);
}
